// GCBlock2Torch_5196910428401
// MI455X (gfx1250) — matrix operand order repaired and checked by reading, not compiled, not run on hardware
//
#include <hip/hip_runtime.h>
#include <hip/hip_bf16.h>

#define N_ATOMS 20000
#define N_PAIRS 320000
#define N_BASIS 10
#define CDIM    128
#define PI_W    64

typedef __attribute__((ext_vector_type(16))) __bf16 v16bf;
typedef __attribute__((ext_vector_type(8)))  float  v8f;

union FragAB {
    unsigned d[8];
    uint4    u[2];
    v16bf    v;
};

// ---------- bf16 helpers (round-to-nearest-even) ----------
__device__ __forceinline__ unsigned short f2bf(float x) {
    unsigned u = __float_as_uint(x);
    unsigned r = (u + 0x7fffu + ((u >> 16) & 1u)) >> 16;
    return (unsigned short)r;
}
__device__ __forceinline__ float bf2f(unsigned short h) {
    return __uint_as_float(((unsigned)h) << 16);
}
__device__ __forceinline__ unsigned pack2f(float a, float b) {
    return (unsigned)f2bf(a) | ((unsigned)f2bf(b) << 16);
}
__device__ __forceinline__ float fast_tanh(float x) {
    // tanh(x) = 1 - 2/(exp2(2x*log2e)+1) -> v_exp_f32 + v_rcp_f32 + 2 FMA
    float e = __builtin_amdgcn_exp2f(x * 2.8853900817779268f);
    return 1.0f - 2.0f * __builtin_amdgcn_rcpf(e + 1.0f);
}

__device__ __forceinline__ v8f wmma_bf16(const FragAB& a, const FragAB& b, v8f c) {
    return __builtin_amdgcn_wmma_f32_16x16x32_bf16(false, a.v, false, b.v,
                                                   (short)0, c, false, false);
}

// B fragment: pre-swizzled weights, ((ks*NT+nt)*32+lane)*8 dwords, 2x uint4
__device__ __forceinline__ FragAB loadB(const unsigned* __restrict__ W,
                                        int ks, int nt, int NT, int lane) {
    FragAB f;
    const uint4* p = (const uint4*)(W + (((size_t)(ks * NT + nt) * 32 + lane) << 3));
    f.u[0] = p[0];
    f.u[1] = p[1];
    return f;
}

// A fragment from a bf16 row (global or LDS). 16-bit A 16x32 layout:
// lanes<16: K 0..7 (bytes 0..15) and 16..23 (bytes 32..47); lanes>=16: +8 K (+16B)
__device__ __forceinline__ FragAB loadA_bf(const unsigned short* row, int ks, int laneHi) {
    const char* base = (const char*)row + ks * 64 + laneHi * 16;
    FragAB f;
    f.u[0] = *(const uint4*)(base);
    f.u[1] = *(const uint4*)(base + 32);
    return f;
}

// A fragment from an f32 row, converting to bf16 on the fly
__device__ __forceinline__ FragAB loadA_f32(const float* row, int ks, int laneHi) {
    FragAB f;
#pragma unroll
    for (int v = 0; v < 8; ++v) {
        int koff = (v < 4 ? 2 * v : 16 + 2 * (v - 4)) + laneHi * 8;
        const float2 w = *(const float2*)(row + ks * 32 + koff);
        f.d[v] = pack2f(w.x, w.y);
    }
    return f;
}

// C/D tile: VGPR v, lane l -> row = v + 8*(l>=16), col = colBase + (l&15)
__device__ __forceinline__ void store_tile_tanh_bf16(unsigned short* buf, int stride,
                                                     v8f acc, int colBase, int lane) {
    int laneHi = lane >> 4;
    int col = colBase + (lane & 15);
#pragma unroll
    for (int v = 0; v < 8; ++v)
        buf[(v + 8 * laneHi) * stride + col] = f2bf(fast_tanh(acc[v]));
}
__device__ __forceinline__ void store_tile_bf16(unsigned short* buf, int stride,
                                                v8f acc, int colBase, int lane) {
    int laneHi = lane >> 4;
    int col = colBase + (lane & 15);
#pragma unroll
    for (int v = 0; v < 8; ++v)
        buf[(v + 8 * laneHi) * stride + col] = f2bf(acc[v]);
}

__device__ __forceinline__ void atomic_add_f32(float* p, float v) {
    __hip_atomic_fetch_add(p, v, __ATOMIC_RELAXED, __HIP_MEMORY_SCOPE_AGENT);
}

// ============================================================================
// Weight pack: f32 row-major (Ktot x Ntot) -> bf16 WMMA-B fragment order
// ============================================================================
__global__ void pack_wmma_b(const float* __restrict__ src, unsigned* __restrict__ dst,
                            int Ktot, int Ntot) {
    int total = (Ktot * Ntot) >> 1;
    int NT = Ntot >> 4;
    for (int d = blockIdx.x * blockDim.x + threadIdx.x; d < total;
         d += gridDim.x * blockDim.x) {
        int v = d & 7;
        int l = (d >> 3) & 31;
        int f = d >> 8;
        int nt = f % NT;
        int kt = f / NT;
        int n = nt * 16 + (l & 15);
        int k = kt * 32 + ((l >> 4) << 3) + (v < 4 ? 2 * v : 16 + 2 * (v - 4));
        dst[d] = pack2f(src[(size_t)k * Ntot + n], src[(size_t)(k + 1) * Ntot + n]);
    }
}

// ============================================================================
// Kernel 1: p1_in = tanh(tanh(p1 @ Wpre0 + b0) @ Wpre1 + b1)  -> bf16
// 2 waves/block, 16 atoms/wave, 625 blocks (exact)
// ============================================================================
#define K1_WAVES 2
__global__ void k1_premlp(const float* __restrict__ p1,
                          const float* __restrict__ b0, const float* __restrict__ b1,
                          const unsigned* __restrict__ Wp0, const unsigned* __restrict__ Wp1,
                          unsigned short* __restrict__ p1in) {
    __shared__ __align__(16) unsigned short bufA[K1_WAVES][16 * 128];
    __shared__ __align__(16) unsigned short bufB[K1_WAVES][16 * 128];
    const int tid = threadIdx.x, wave = tid >> 5, lane = tid & 31;
    const int laneHi = lane >> 4, rl = lane & 15;
    const int atomBase = (blockIdx.x * K1_WAVES + wave) * 16;

    const float* rowP = p1 + (size_t)(atomBase + rl) * CDIM;
    FragAB aF[4];
#pragma unroll
    for (int ks = 0; ks < 4; ++ks) aF[ks] = loadA_f32(rowP, ks, laneHi);

    for (int nt = 0; nt < 8; ++nt) {
        float bias = b0[nt * 16 + rl];
        v8f acc;
#pragma unroll
        for (int v = 0; v < 8; ++v) acc[v] = bias;
#pragma unroll
        for (int ks = 0; ks < 4; ++ks) acc = wmma_bf16(aF[ks], loadB(Wp0, ks, nt, 8, lane), acc);
        store_tile_tanh_bf16(bufA[wave], 128, acc, nt * 16, lane);
    }
    __syncthreads();

    const unsigned short* rA = &bufA[wave][rl * 128];
    FragAB aG[4];
#pragma unroll
    for (int ks = 0; ks < 4; ++ks) aG[ks] = loadA_bf(rA, ks, laneHi);
    for (int nt = 0; nt < 8; ++nt) {
        float bias = b1[nt * 16 + rl];
        v8f acc;
#pragma unroll
        for (int v = 0; v < 8; ++v) acc[v] = bias;
#pragma unroll
        for (int ks = 0; ks < 4; ++ks) acc = wmma_bf16(aG[ks], loadB(Wp1, ks, nt, 8, lane), acc);
        store_tile_tanh_bf16(bufB[wave], 128, acc, nt * 16, lane);
    }
    __syncthreads();

    // coalesced bf16 row copy: 16 atoms x 64 dwords
    const unsigned* s = (const unsigned*)bufB[wave];
    unsigned* dst = (unsigned*)p1in + (size_t)atomBase * 64;
#pragma unroll
    for (int t = 0; t < 8; ++t) {
        int di = t * 128 + lane * 4;
        *(uint4*)(dst + di) = *(const uint4*)(s + di);
    }
}

// ============================================================================
// Kernel 2: fused pair chain (GEMM 256->640 + tanh + basis + 64->128 + 128->256,
//           atomic segment sums into p1acc / p3acc)
// 4 waves/block, 16 pairs/wave, 5000 blocks (exact)
// ============================================================================
#define K2_WAVES 4
__global__ void k2_pairs(const int* __restrict__ ind2,
                         const unsigned short* __restrict__ p1in,
                         const float* __restrict__ p3, const float* __restrict__ d3,
                         const float* __restrict__ basis,
                         const unsigned* __restrict__ Wpi, const float* __restrict__ bpi,
                         const unsigned* __restrict__ Wii0, const unsigned* __restrict__ Wii1,
                         float* __restrict__ p1acc, float* __restrict__ p3acc) {
    __shared__ __align__(16) unsigned short hbuf[K2_WAVES][16 * 80];
    __shared__ __align__(16) unsigned short ipair[K2_WAVES][16 * 64];
    __shared__ __align__(16) unsigned short ybuf[K2_WAVES][16 * 128];
    __shared__ float baswS[K2_WAVES][16][N_BASIS];
    __shared__ float coefS[K2_WAVES][16][3];
    __shared__ int   idxiS[K2_WAVES][16];
    const int tid = threadIdx.x, wave = tid >> 5, lane = tid & 31;
    const int laneHi = lane >> 4, rl = lane & 15;
    const int pairBase = (blockIdx.x * K2_WAVES + wave) * 16;

    if (lane < 16) {
        int p = pairBase + lane;
        int ia = ind2[p * 2 + 0];
        int ja = ind2[p * 2 + 1];
        idxiS[wave][lane] = ia;
#pragma unroll
        for (int x = 0; x < 3; ++x)
            coefS[wave][lane][x] = p3[(size_t)ja * 3 + x] + d3[(size_t)p * 3 + x];
#pragma unroll
        for (int b = 0; b < N_BASIS; ++b)
            baswS[wave][lane][b] = basis[(size_t)p * N_BASIS + b];
    }
    __syncthreads();

    // Gathered A rows (bf16, 256B per atom row). concat: ks<4 -> i side, ks>=4 -> j side
    int p = pairBase + rl;
    int ia = ind2[p * 2 + 0];
    int ja = ind2[p * 2 + 1];
    const unsigned short* rowI = p1in + (size_t)ia * CDIM;
    const unsigned short* rowJ = p1in + (size_t)ja * CDIM;
    FragAB aF[8];
#pragma unroll
    for (int ks = 0; ks < 8; ++ks)
        aF[ks] = loadA_bf(ks < 4 ? rowI : rowJ, ks & 3, laneHi);

    // ---- Phase A: h = tanh(x @ W_pi + b_pi) in 8 chunks of 80 cols,
    //               fused basis contraction -> ipair (16x64 bf16)
    for (int ch = 0; ch < 8; ++ch) {
        for (int ntl = 0; ntl < 5; ++ntl) {
            int nt = ch * 5 + ntl;
            float bias = bpi[nt * 16 + rl];
            v8f acc;
#pragma unroll
            for (int v = 0; v < 8; ++v) acc[v] = bias;
#pragma unroll
            for (int ks = 0; ks < 8; ++ks)
                acc = wmma_bf16(aF[ks], loadB(Wpi, ks, nt, 40, lane), acc);
            int col = ntl * 16 + rl;
#pragma unroll
            for (int v = 0; v < 8; ++v)
                hbuf[wave][(v + 8 * laneHi) * 80 + col] = f2bf(fast_tanh(acc[v]));
        }
        __syncthreads();
#pragma unroll
        for (int t = 0; t < 4; ++t) {
            int o = t * 32 + lane;
            int r = o >> 3, cg = o & 7;
            float s = 0.0f;
#pragma unroll
            for (int b = 0; b < N_BASIS; ++b)
                s += bf2f(hbuf[wave][r * 80 + cg * N_BASIS + b]) * baswS[wave][r][b];
            ipair[wave][r * 64 + ch * 8 + cg] = f2bf(s);
        }
        __syncthreads();
    }

    // ---- Phase B: y = tanh(ipair @ W_ii0), 64->128
    const unsigned short* rIp = &ipair[wave][rl * 64];
    FragAB aB[2];
#pragma unroll
    for (int ks = 0; ks < 2; ++ks) aB[ks] = loadA_bf(rIp, ks, laneHi);
    for (int nt = 0; nt < 8; ++nt) {
        v8f acc = {};
#pragma unroll
        for (int ks = 0; ks < 2; ++ks)
            acc = wmma_bf16(aB[ks], loadB(Wii0, ks, nt, 8, lane), acc);
        store_tile_tanh_bf16(ybuf[wave], 128, acc, nt * 16, lane);
    }
    __syncthreads();

    // per-lane row caches for the scatter
    float cf[8][3];
    int ai[8];
#pragma unroll
    for (int v = 0; v < 8; ++v) {
        int r = v + 8 * laneHi;
        ai[v] = idxiS[wave][r];
        cf[v][0] = coefS[wave][r][0];
        cf[v][1] = coefS[wave][r][1];
        cf[v][2] = coefS[wave][r][2];
    }

    // ---- Phase C: z = tanh(y @ W_ii1), 128->256, scatter-add
    const unsigned short* rY = &ybuf[wave][rl * 128];
    FragAB aC[4];
#pragma unroll
    for (int ks = 0; ks < 4; ++ks) aC[ks] = loadA_bf(rY, ks, laneHi);
    for (int nt = 0; nt < 16; ++nt) {
        v8f acc = {};
#pragma unroll
        for (int ks = 0; ks < 4; ++ks)
            acc = wmma_bf16(aC[ks], loadB(Wii1, ks, nt, 16, lane), acc);
        int col = nt * 16 + rl;
#pragma unroll
        for (int v = 0; v < 8; ++v) {
            float z = fast_tanh(acc[v]);
            atomic_add_f32(p1acc + (size_t)ai[v] * 256 + col, z);
            if (nt >= 8) {
                int c = col - 128;
#pragma unroll
                for (int x = 0; x < 3; ++x)
                    atomic_add_f32(p3acc + (size_t)ai[v] * 384 + x * 128 + c, z * cf[v][x]);
            }
        }
    }
}

// ============================================================================
// Kernel 3: atom post: W_po0/W_po1, W_eq on (atom,xyz), dotted, W_t0/W_t1, gate
// 2 waves/block, 16 atoms/wave, 625 blocks (exact)
// ============================================================================
#define K3_WAVES 2
__global__ void k3_post(const float* __restrict__ p1acc, const float* __restrict__ p3acc,
                        const unsigned* __restrict__ Wpo0, const unsigned* __restrict__ Wpo1,
                        const unsigned* __restrict__ Weq,
                        const unsigned* __restrict__ Wt0, const float* __restrict__ bt0,
                        const unsigned* __restrict__ Wt1, const float* __restrict__ bt1,
                        float* __restrict__ p1out, float* __restrict__ p3out) {
    __shared__ __align__(16) unsigned short bufA[K3_WAVES][16 * 128];  // u -> dotted
    __shared__ __align__(16) unsigned short bufB[K3_WAVES][16 * 128];  // p1_new -> s3
    __shared__ __align__(16) unsigned short sbuf[K3_WAVES][16 * 128];  // t
    __shared__ __align__(16) unsigned short p3n[K3_WAVES][48 * 128];   // p3_new
    const int tid = threadIdx.x, wave = tid >> 5, lane = tid & 31;
    const int laneHi = lane >> 4, rl = lane & 15;
    const int atomBase = (blockIdx.x * K3_WAVES + wave) * 16;

    // G1: u = tanh(p1acc @ W_po0)  (256 -> 128)
    {
        const float* rowAcc = p1acc + (size_t)(atomBase + rl) * 256;
        FragAB aF[8];
#pragma unroll
        for (int ks = 0; ks < 8; ++ks) aF[ks] = loadA_f32(rowAcc, ks, laneHi);
        for (int nt = 0; nt < 8; ++nt) {
            v8f acc = {};
#pragma unroll
            for (int ks = 0; ks < 8; ++ks)
                acc = wmma_bf16(aF[ks], loadB(Wpo0, ks, nt, 8, lane), acc);
            store_tile_tanh_bf16(bufA[wave], 128, acc, nt * 16, lane);
        }
    }
    __syncthreads();

    // G2: p1_new = tanh(u @ W_po1) -> bufB
    {
        const unsigned short* rA = &bufA[wave][rl * 128];
        FragAB aF[4];
#pragma unroll
        for (int ks = 0; ks < 4; ++ks) aF[ks] = loadA_bf(rA, ks, laneHi);
        for (int nt = 0; nt < 8; ++nt) {
            v8f acc = {};
#pragma unroll
            for (int ks = 0; ks < 4; ++ks)
                acc = wmma_bf16(aF[ks], loadB(Wpo1, ks, nt, 8, lane), acc);
            store_tile_tanh_bf16(bufB[wave], 128, acc, nt * 16, lane);
        }
    }
    __syncthreads();

    // G3: p3_new = p3acc @ W_eq over 48 (atom,x) rows -> p3n (bf16, no tanh)
    for (int rt = 0; rt < 3; ++rt) {
        const float* rowP3 = p3acc + (size_t)(atomBase * 3 + rt * 16 + rl) * 128;
        FragAB aF[4];
#pragma unroll
        for (int ks = 0; ks < 4; ++ks) aF[ks] = loadA_f32(rowP3, ks, laneHi);
        for (int nt = 0; nt < 8; ++nt) {
            v8f acc = {};
#pragma unroll
            for (int ks = 0; ks < 4; ++ks)
                acc = wmma_bf16(aF[ks], loadB(Weq, ks, nt, 8, lane), acc);
            int col = nt * 16 + rl;
#pragma unroll
            for (int v = 0; v < 8; ++v)
                p3n[wave][(rt * 16 + v + 8 * laneHi) * 128 + col] = f2bf(acc[v]);
        }
    }
    __syncthreads();

    // dotted[a][c] = sum_x p3_new[a][x][c]^2 -> bufA (reuse)
    for (int t = 0; t < 64; ++t) {
        int o = t * 32 + lane;
        int r = o >> 7, c = o & 127;
        float d = 0.0f;
#pragma unroll
        for (int x = 0; x < 3; ++x) {
            float w = bf2f(p3n[wave][(r * 3 + x) * 128 + c]);
            d += w * w;
        }
        bufA[wave][o] = f2bf(d);
    }
    __syncthreads();

    // G4: t = tanh([p1_new | dotted] @ W_t0 + b_t0) -> sbuf
    {
        const unsigned short* rB = &bufB[wave][rl * 128];
        const unsigned short* rD = &bufA[wave][rl * 128];
        FragAB aF[8];
#pragma unroll
        for (int ks = 0; ks < 8; ++ks)
            aF[ks] = (ks < 4) ? loadA_bf(rB, ks, laneHi) : loadA_bf(rD, ks - 4, laneHi);
        for (int nt = 0; nt < 8; ++nt) {
            float bias = bt0[nt * 16 + rl];
            v8f acc;
#pragma unroll
            for (int v = 0; v < 8; ++v) acc[v] = bias;
#pragma unroll
            for (int ks = 0; ks < 8; ++ks)
                acc = wmma_bf16(aF[ks], loadB(Wt0, ks, nt, 8, lane), acc);
            store_tile_tanh_bf16(sbuf[wave], 128, acc, nt * 16, lane);
        }
    }
    __syncthreads();

    // G5: t2 = tanh(t @ W_t1 + b_t1); nt<8 -> p1_out; nt>=8 -> s3 (bufB, reuse)
    {
        const unsigned short* rT = &sbuf[wave][rl * 128];
        FragAB aF[4];
#pragma unroll
        for (int ks = 0; ks < 4; ++ks) aF[ks] = loadA_bf(rT, ks, laneHi);
        for (int nt = 0; nt < 16; ++nt) {
            float bias = bt1[nt * 16 + rl];
            v8f acc;
#pragma unroll
            for (int v = 0; v < 8; ++v) acc[v] = bias;
#pragma unroll
            for (int ks = 0; ks < 4; ++ks)
                acc = wmma_bf16(aF[ks], loadB(Wt1, ks, nt, 16, lane), acc);
            int col = nt * 16 + rl;
#pragma unroll
            for (int v = 0; v < 8; ++v) {
                float t2 = fast_tanh(acc[v]);
                int row = v + 8 * laneHi;
                if (nt < 8)
                    p1out[(size_t)(atomBase + row) * 128 + col] = t2;
                else
                    bufB[wave][row * 128 + (col - 128)] = f2bf(t2);
            }
        }
    }
    __syncthreads();

    // p3_out = p3_new * s3, coalesced float4 stores
    float* dst = p3out + (size_t)atomBase * 384;
    for (int q = 0; q < 48; ++q) {
        int o4 = q * 32 + lane;
        int f = o4 * 4;
        int g = f >> 7;        // local (atom,x) row 0..47
        int la = g / 3;        // local atom
        int cb = f & 127;
        float4 r;
        r.x = bf2f(p3n[wave][g * 128 + cb + 0]) * bf2f(bufB[wave][la * 128 + cb + 0]);
        r.y = bf2f(p3n[wave][g * 128 + cb + 1]) * bf2f(bufB[wave][la * 128 + cb + 1]);
        r.z = bf2f(p3n[wave][g * 128 + cb + 2]) * bf2f(bufB[wave][la * 128 + cb + 2]);
        r.w = bf2f(p3n[wave][g * 128 + cb + 3]) * bf2f(bufB[wave][la * 128 + cb + 3]);
        *(float4*)(dst + f) = r;
    }
}

// ============================================================================
// Host launcher
// ============================================================================
extern "C" void kernel_launch(void* const* d_in, const int* in_sizes, int n_in,
                              void* d_out, int out_size, void* d_ws, size_t ws_size,
                              hipStream_t stream) {
    const int*   ind2  = (const int*)d_in[0];
    const float* p1    = (const float*)d_in[1];
    const float* p3    = (const float*)d_in[2];
    const float* d3    = (const float*)d_in[3];
    const float* basis = (const float*)d_in[4];
    const float* Wpre0 = (const float*)d_in[5];
    const float* bpre0 = (const float*)d_in[6];
    const float* Wpre1 = (const float*)d_in[7];
    const float* bpre1 = (const float*)d_in[8];
    const float* Wpi   = (const float*)d_in[9];
    const float* bpi   = (const float*)d_in[10];
    const float* Wii0  = (const float*)d_in[11];
    const float* Wii1  = (const float*)d_in[12];
    const float* Wpo0  = (const float*)d_in[13];
    const float* Wpo1  = (const float*)d_in[14];
    const float* Weq   = (const float*)d_in[15];
    const float* Wt0   = (const float*)d_in[16];
    const float* bt0   = (const float*)d_in[17];
    const float* Wt1   = (const float*)d_in[18];
    const float* bt1   = (const float*)d_in[19];

    float* p1out = (float*)d_out;
    float* p3out = p1out + (size_t)N_ATOMS * 128;

    // workspace carve-out (256B aligned)
    char* ws = (char*)d_ws;
    size_t off = 0;
    auto carve = [&](size_t bytes) -> void* {
        void* p = ws + off;
        off += (bytes + 255) & ~(size_t)255;
        return p;
    };
    unsigned short* p1in  = (unsigned short*)carve((size_t)N_ATOMS * 128 * 2);
    float*          p1acc = (float*)carve((size_t)N_ATOMS * 256 * 4);
    float*          p3acc = (float*)carve((size_t)N_ATOMS * 384 * 4);
    unsigned* Wpre0s = (unsigned*)carve((size_t)128 * 128 * 2);
    unsigned* Wpre1s = (unsigned*)carve((size_t)128 * 128 * 2);
    unsigned* Wpis   = (unsigned*)carve((size_t)256 * 640 * 2);
    unsigned* Wii0s  = (unsigned*)carve((size_t)64 * 128 * 2);
    unsigned* Wii1s  = (unsigned*)carve((size_t)128 * 256 * 2);
    unsigned* Wpo0s  = (unsigned*)carve((size_t)256 * 128 * 2);
    unsigned* Wpo1s  = (unsigned*)carve((size_t)128 * 128 * 2);
    unsigned* Weqs   = (unsigned*)carve((size_t)128 * 128 * 2);
    unsigned* Wt0s   = (unsigned*)carve((size_t)256 * 128 * 2);
    unsigned* Wt1s   = (unsigned*)carve((size_t)128 * 256 * 2);

    hipMemsetAsync(p1acc, 0, (size_t)N_ATOMS * 256 * 4, stream);
    hipMemsetAsync(p3acc, 0, (size_t)N_ATOMS * 384 * 4, stream);

    auto pack = [&](const float* src, unsigned* dst, int K, int N) {
        int total = K * N / 2;
        int blocks = (total + 255) / 256;
        hipLaunchKernelGGL(pack_wmma_b, dim3(blocks), dim3(256), 0, stream, src, dst, K, N);
    };
    pack(Wpre0, Wpre0s, 128, 128);
    pack(Wpre1, Wpre1s, 128, 128);
    pack(Wpi,   Wpis,   256, 640);
    pack(Wii0,  Wii0s,  64,  128);
    pack(Wii1,  Wii1s,  128, 256);
    pack(Wpo0,  Wpo0s,  256, 128);
    pack(Wpo1,  Wpo1s,  128, 128);
    pack(Weq,   Weqs,   128, 128);
    pack(Wt0,   Wt0s,   256, 128);
    pack(Wt1,   Wt1s,   128, 256);

    hipLaunchKernelGGL(k1_premlp, dim3(N_ATOMS / (16 * K1_WAVES)), dim3(32 * K1_WAVES), 0, stream,
                       p1, bpre0, bpre1, Wpre0s, Wpre1s, p1in);
    hipLaunchKernelGGL(k2_pairs, dim3(N_PAIRS / (16 * K2_WAVES)), dim3(32 * K2_WAVES), 0, stream,
                       ind2, p1in, p3, d3, basis, Wpis, bpi, Wii0s, Wii1s, p1acc, p3acc);
    hipLaunchKernelGGL(k3_post, dim3(N_ATOMS / (16 * K3_WAVES)), dim3(32 * K3_WAVES), 0, stream,
                       p1acc, p3acc, Wpo0s, Wpo1s, Weqs, Wt0s, bt0, Wt1s, bt1, p1out, p3out);
}
